// AttentionMILPooling_8787503088263
// MI455X (gfx1250) — matrix operand order repaired and checked by reading, not compiled, not run on hardware
//
#include <hip/hip_runtime.h>

// ---------------------------------------------------------------------------
// AttentionMILPooling for MI455X (gfx1250).
// Phase 1 (bf16 WMMA GEMM): scores = tanh(X@W1+b1)@W2+b2, global max
// Phase 2: sum of exp(s-max)
// Phase 3: per-bag weighted segment sum of X rows
// HBM-bound design: ~537 MB total traffic => ~23us floor at 23.3 TB/s.
// ---------------------------------------------------------------------------

#define TTOT   131072
#define FDIM   512
#define HDIM   256
#define NBAGS  64
#define W1T_STRIDE 528                       // bf16 elems per row (512 + 16 pad)
#define W1T_ELEMS  (HDIM * W1T_STRIDE)       // 135168
#define W1T_BYTES  (W1T_ELEMS * 2)           // 270336 bytes (fits 320KB LDS)

typedef __attribute__((ext_vector_type(16))) __bf16 v16bf;
typedef __attribute__((ext_vector_type(8)))  float  v8f;
typedef __attribute__((ext_vector_type(8)))  float  f8;
typedef int v4i __attribute__((vector_size(16)));   // matches builtin's V4i

#if defined(__HIP_DEVICE_COMPILE__) && \
    __has_builtin(__builtin_amdgcn_global_load_async_to_lds_b128) && \
    __has_builtin(__builtin_amdgcn_s_wait_asynccnt)
#define HAVE_ASYNC_LDS 1
#endif

__device__ __forceinline__ void asyncCopy16(void* lds, const void* g) {
#ifdef HAVE_ASYNC_LDS
  __builtin_amdgcn_global_load_async_to_lds_b128(
      (__attribute__((address_space(1))) v4i*)(g),
      (__attribute__((address_space(3))) v4i*)(lds), 0, 0);
#else
  *(uint4*)lds = *(const uint4*)g;
#endif
}

// float atomic max via signed-max / unsigned-min bit trick (init 0xFF800000)
__device__ __forceinline__ void atomicMaxF(float* a, float v) {
  if (v >= 0.0f) atomicMax((int*)a, __float_as_int(v));
  else           atomicMin((unsigned int*)a, __float_as_uint(v));
}

// ---------------------------------------------------------------------------
// K0: zero output, reset softmax cells, build padded bf16 W1^T in workspace.
// w1t_g[n*528 + k] = bf16(W1[k][n]); pad cols 512..527 = 0.
// ---------------------------------------------------------------------------
__global__ void kInit(const float* __restrict__ W1, __bf16* __restrict__ w1t_g,
                      float* __restrict__ out, float* __restrict__ wsHead) {
  const int idx = blockIdx.x * blockDim.x + threadIdx.x;
  if (idx < W1T_ELEMS) {
    const int n = idx / W1T_STRIDE, k = idx % W1T_STRIDE;
    const float v = (k < FDIM) ? W1[k * HDIM + n] : 0.0f;
    w1t_g[idx] = (__bf16)v;
  }
  if (idx < NBAGS * FDIM) out[idx] = 0.0f;
  if (idx == 0) {
    ((unsigned int*)wsHead)[0] = 0xFF800000u;  // -inf
    wsHead[1] = 0.0f;                          // sum(exp)
  }
}

// ---------------------------------------------------------------------------
// K1: scores via bf16 WMMA. 8 waves/block, wave owns a 16-row tile.
// LDS: W1^T bf16 [256][528] (264KB) + b1 + W2.
// Double-buffered: B fragments across the n-loop (keeps ds loads in flight
// past each WMMA -> s_wait_dscnt <= 2 instead of 0), A chunks across ks.
// ---------------------------------------------------------------------------
__global__ void __launch_bounds__(256)
kScores(const float* __restrict__ X, const __bf16* __restrict__ w1t_g,
        const float* __restrict__ b1, const float* __restrict__ W2,
        const float* __restrict__ b2, float* __restrict__ scores,
        float* __restrict__ wsHead, int numChunks) {
  extern __shared__ __align__(16) char smem[];
  __bf16* w1t = (__bf16*)smem;
  float*  b1s = (float*)(smem + W1T_BYTES);
  float*  w2s = b1s + HDIM;

  const int t = threadIdx.x;

  // Stage W1^T into LDS (async global->LDS, 16896 b128 transfers / block).
  {
    const char* g = (const char*)w1t_g;
    char* l = (char*)smem;
    for (int ofs = t * 16; ofs < W1T_BYTES; ofs += 256 * 16)
      asyncCopy16(l + ofs, g + ofs);
#ifdef HAVE_ASYNC_LDS
    __builtin_amdgcn_s_wait_asynccnt(0);
#endif
    if (t < HDIM) { b1s[t] = b1[t]; w2s[t] = W2[t]; }
  }
  __syncthreads();

  const int wave = t >> 5;
  const int lane = t & 31;
  const int h = lane >> 4;   // K half (A) / K half (B)
  const int m = lane & 15;   // A: row in tile;  B/C: column in tile
  const float bias2 = b2[0];

  for (int chunk = blockIdx.x; chunk < numChunks; chunk += gridDim.x) {
    const int rowBase = chunk * 128 + wave * 16;
    const f8* ap = (const f8*)(X + (size_t)(rowBase + m) * FDIM);

    v8f acc[16];
#pragma unroll
    for (int n = 0; n < 16; ++n) {
      v8f z = {0.f, 0.f, 0.f, 0.f, 0.f, 0.f, 0.f, 0.f};
      acc[n] = z;
    }

    // A fragment (16x32 bf16): lane(h,m) holds row m, K = ks*32 + h*16+{0..15}
    // (two contiguous float8 groups).
    f8 g0 = ap[2 * h];
    f8 g1 = ap[2 * h + 1];

    for (int ks = 0; ks < 16; ++ks) {
      v16bf a;
#pragma unroll
      for (int i = 0; i < 8; ++i) { a[i] = (__bf16)g0[i]; a[i + 8] = (__bf16)g1[i]; }

      // issue next ks's global A loads early (hidden behind 16 WMMAs)
      f8 ng0 = g0, ng1 = g1;
      if (ks < 15) {
        ng0 = ap[(ks + 1) * 4 + 2 * h];
        ng1 = ap[(ks + 1) * 4 + 2 * h + 1];
      }

      // B fragment (32x16 bf16): lane = column n*16+m, 16 contiguous K values
      // starting at ks*32 + h*16 (one 32B LDS read); double-buffered over n.
      const __bf16* bbase = w1t + (size_t)m * W1T_STRIDE + ks * 32 + h * 16;
      v16bf bcur = *(const v16bf*)(bbase);
#pragma unroll
      for (int n = 0; n < 16; ++n) {
        v16bf bnext = bcur;
        if (n < 15)
          bnext = *(const v16bf*)(bbase + (size_t)(n + 1) * (16 * W1T_STRIDE));
        acc[n] = __builtin_amdgcn_wmma_f32_16x16x32_bf16(
            false, a, false, bcur, (short)0, acc[n], false, false);
        bcur = bnext;
      }
      g0 = ng0; g1 = ng1;
    }

    // Epilogue: s[M] = sum_N tanh(C[M][N] + b1[N]) * W2[N] + b2.
    // C layout: lane(h,m) VGPR v -> (M = v + 8h, N = m) within tile.
    float part[8];
#pragma unroll
    for (int v = 0; v < 8; ++v) part[v] = 0.0f;
#pragma unroll
    for (int n = 0; n < 16; ++n) {
      const int nn = n * 16 + m;
      const float bb = b1s[nn], ww = w2s[nn];
#pragma unroll
      for (int v = 0; v < 8; ++v) part[v] += tanhf(acc[n][v] + bb) * ww;
    }
    // reduce over the 16 lanes sharing the same h (xor of low 4 lane bits)
#pragma unroll
    for (int mask = 1; mask <= 8; mask <<= 1) {
#pragma unroll
      for (int v = 0; v < 8; ++v) part[v] += __shfl_xor(part[v], mask, 32);
    }
    if (m == 0) {
#pragma unroll
      for (int v = 0; v < 8; ++v) {
        const float s = part[v] + bias2;
        scores[rowBase + h * 8 + v] = s;
        atomicMaxF(wsHead, s);
      }
    }
  }
}

// ---------------------------------------------------------------------------
// K2: sum of exp(score - max) -> wsHead[1]
// ---------------------------------------------------------------------------
__global__ void kSumExp(const float* __restrict__ scores, float* __restrict__ wsHead) {
  const int i = blockIdx.x * blockDim.x + threadIdx.x;
  const float gmax = wsHead[0];
  float v = (i < TTOT) ? expf(scores[i] - gmax) : 0.0f;
#pragma unroll
  for (int mask = 16; mask >= 1; mask >>= 1) v += __shfl_xor(v, mask, 32);
  if ((threadIdx.x & 31) == 0) atomicAdd(&wsHead[1], v);
}

// ---------------------------------------------------------------------------
// K3: out[bag] += w_i * X[i].  Block = 128 rows; 2 groups of 128 threads,
// thread owns one float4 column, register accumulate, flush on bag change.
// ---------------------------------------------------------------------------
__global__ void __launch_bounds__(256)
kSegSum(const float* __restrict__ X, const float* __restrict__ scores,
        const int* __restrict__ bagSizes, const float* __restrict__ wsHead,
        float* __restrict__ out) {
  __shared__ int cum[NBAGS];
  __shared__ float wrow[128];
  const int t = threadIdx.x;
  if (t < NBAGS) {                 // inclusive prefix of ragged bag sizes
    int c = 0;
    for (int j = 0; j <= t; ++j) c += bagSizes[j];
    cum[t] = c;
  }
  const int rowBase = blockIdx.x * 128;
  const float gmax = wsHead[0];
  const float inv  = 1.0f / wsHead[1];
  if (t < 128) wrow[t] = expf(scores[rowBase + t] - gmax) * inv;
  __syncthreads();

  const int g = t >> 7, c = t & 127;
  const int r0 = g * 64, r1 = r0 + 64;
  int bag = 0;
  { const int i0 = rowBase + r0; while (bag < NBAGS - 1 && cum[bag] <= i0) ++bag; }
  int curBag = bag;
  float4 acc = make_float4(0.f, 0.f, 0.f, 0.f);
  for (int r = r0; r < r1; ++r) {
    const int i = rowBase + r;
    while (bag < NBAGS - 1 && cum[bag] <= i) ++bag;   // searchsorted(right)
    if (bag != curBag) {
      float* o = out + (size_t)curBag * FDIM + c * 4;
      atomicAdd(o + 0, acc.x); atomicAdd(o + 1, acc.y);
      atomicAdd(o + 2, acc.z); atomicAdd(o + 3, acc.w);
      acc = make_float4(0.f, 0.f, 0.f, 0.f);
      curBag = bag;
    }
    const float w = wrow[r];
    const float4 x4 = ((const float4*)(X + (size_t)i * FDIM))[c];
    acc.x += w * x4.x; acc.y += w * x4.y;
    acc.z += w * x4.z; acc.w += w * x4.w;
  }
  float* o = out + (size_t)curBag * FDIM + c * 4;
  atomicAdd(o + 0, acc.x); atomicAdd(o + 1, acc.y);
  atomicAdd(o + 2, acc.z); atomicAdd(o + 3, acc.w);
}

// ---------------------------------------------------------------------------
extern "C" void kernel_launch(void* const* d_in, const int* in_sizes, int n_in,
                              void* d_out, int out_size, void* d_ws, size_t ws_size,
                              hipStream_t stream) {
  (void)in_sizes; (void)n_in; (void)out_size; (void)ws_size;
  const float* X        = (const float*)d_in[0];
  const int*   bagSizes = (const int*)d_in[1];
  const float* W1       = (const float*)d_in[2];
  const float* b1       = (const float*)d_in[3];
  const float* W2       = (const float*)d_in[4];
  const float* b2       = (const float*)d_in[5];
  float* out = (float*)d_out;

  // workspace layout: [0..63] softmax max/sum header, then padded bf16 W1^T,
  // then scores[T].  Total ~795 KB.
  float*  wsHead = (float*)d_ws;
  __bf16* w1t_g  = (__bf16*)((char*)d_ws + 64);
  float*  scores = (float*)((char*)d_ws + 64 + W1T_BYTES);

  kInit<<<(W1T_ELEMS + 255) / 256, 256, 0, stream>>>(W1, w1t_g, out, wsHead);

  const size_t ldsBytes = W1T_BYTES + 2 * HDIM * sizeof(float);  // 272384 < 320KB
  kScores<<<512, 256, ldsBytes, stream>>>(X, w1t_g, b1, W2, b2, scores, wsHead,
                                          TTOT / 128);

  kSumExp<<<TTOT / 256, 256, 0, stream>>>(scores, wsHead);

  kSegSum<<<TTOT / 128, 256, 0, stream>>>(X, scores, bagSizes, wsHead, out);
}
